// WPDCPoseLoss_9509057593489
// MI455X (gfx1250) — matrix operand order repaired and checked by reading, not compiled, not run on hardware
//
#include <hip/hip_runtime.h>

typedef __attribute__((ext_vector_type(16))) _Float16 v16h;
typedef __attribute__((ext_vector_type(8)))  _Float16 v8h;
typedef __attribute__((ext_vector_type(8)))  float    v8f;

#define VCOUNT   53215
#define NB       32
#define KD       160                     // padded cols of Y: 3*51 + ones col + pad
#define CHUNK    32                      // K rows (vertices) per LDS slab
#define NCHUNKS  ((VCOUNT + CHUNK - 1) / CHUNK)   // 1663
#define NTILE    10                      // 160/16
#define NPAIRS   55                      // upper-triangle 16x16 tiles
#define GRID     192                     // workgroups for the syrk kernel
#define NLD      20                      // (KD*CHUNK)/256 elements per thread

// workspace layout (float offsets)
#define WS_M     0                       // 160*160 Gram matrix
#define WS_ALPHA (WS_M + KD * KD)        // 32 x 64 (alpha_bar, padded)
#define WS_AVEC  (WS_ALPHA + NB * 64)    // 32 x 12 x 12  (a-vectors)
#define WS_SQD   (WS_AVEC + NB * 144)    // 32 x 12       (inp-target)^2

__device__ __forceinline__ float getM(const float* Mf, int r, int c) {
  // only upper-triangle tiles stored; matrix is symmetric
  return ((r >> 4) <= (c >> 4)) ? Mf[r * KD + c] : Mf[c * KD + r];
}

__global__ void zero_kernel(float* __restrict__ Mf, float* __restrict__ out) {
  int idx = blockIdx.x * blockDim.x + threadIdx.x;
  if (idx < KD * KD) Mf[idx] = 0.0f;
  if (idx == 0) out[0] = 0.0f;
}

__global__ void prep_kernel(const float* __restrict__ inp,
                            const float* __restrict__ target,
                            const float* __restrict__ pmean,
                            const float* __restrict__ pstd,
                            float* __restrict__ ws) {
  const int tid = threadIdx.x;  // 256 threads, 1 block
  float* alpha = ws + WS_ALPHA;
  float* avec  = ws + WS_AVEC;
  float* sqd   = ws + WS_SQD;

  // alpha_bar[n] = [1, tgt[12:52], tgt[52:62], 0-pad] with tgt = std*target+mean
  for (int e = tid; e < NB * 64; e += 256) {
    int n = e >> 6, j = e & 63;
    float val = 0.0f;
    if (j == 0) {
      val = 1.0f;
    } else if (j <= 40) {
      int idx = 12 + (j - 1);
      val = pstd[idx] * target[n * 62 + idx] + pmean[idx];
    } else if (j <= 50) {
      int idx = 52 + (j - 41);
      val = pstd[idx] * target[n * 62 + idx] + pmean[idx];
    }
    alpha[e] = val;
  }
  // a-vec[n][k][j] = (j<=k) ? std[j]*(inp-target)[n,j] : 0   (means cancel)
  for (int e = tid; e < NB * 144; e += 256) {
    int n = e / 144;
    int k = (e / 12) % 12;
    int j = e % 12;
    float val = 0.0f;
    if (j <= k) val = pstd[j] * (inp[n * 12 + j] - target[n * 62 + j]);
    avec[e] = val;
  }
  for (int e = tid; e < NB * 12; e += 256) {
    int n = e / 12, k = e % 12;
    float d = inp[n * 12 + k] - target[n * 62 + k];
    sqd[e] = d * d;
  }
}

// ---- helpers for the SYRK kernel ----

__device__ __forceinline__ void load_regs(int vbase, int tid,
                                          const float* __restrict__ u,
                                          const float* __restrict__ w_shp,
                                          const float* __restrict__ w_exp,
                                          float (&rv)[NLD]) {
#pragma unroll
  for (int i = 0; i < NLD; ++i) {
    int e = tid + 256 * i;
    int col = e % KD;
    int k   = e / KD;
    int v   = vbase + k;
    float val = 0.0f;
    if (v < VCOUNT) {
      if (col < 153) {
        int c = col / 51;
        int j = col - c * 51;
        long row = 3L * v + c;
        if (j == 0)       val = u[row];
        else if (j <= 40) val = w_shp[row * 40 + (j - 1)];
        else              val = w_exp[row * 10 + (j - 41)];
      } else if (col == 153) {
        val = 1.0f;                    // ones column -> sums & count
      }
    }
    rv[i] = val;
  }
}

__device__ __forceinline__ void store_slab(_Float16* __restrict__ buf, int tid,
                                           const float (&rv)[NLD]) {
#pragma unroll
  for (int i = 0; i < NLD; ++i) {
    int e = tid + 256 * i;
    int col = e % KD;
    int k   = e / KD;
    buf[col * CHUNK + k] = (_Float16)rv[i];
  }
}

__device__ __forceinline__ void load_tile(const _Float16* __restrict__ buf,
                                          int mt, int nt, int lane16, int halfSel,
                                          v16h& a, v16h& b) {
  // A tile (16x32, M rows): lane<16 holds K0-7 & K16-23; lane>=16 K8-15 & K24-31
  const _Float16* arow = &buf[(mt * 16 + lane16) * CHUNK + halfSel * 8];
  v8h a0 = *(const v8h*)(arow);
  v8h a1 = *(const v8h*)(arow + 16);
  // B tile (32x16, N cols): lane<16 holds K0-7 & K16-23; lane>=16 K8-15 & K24-31
  const _Float16* bcol = &buf[(nt * 16 + lane16) * CHUNK + halfSel * 8];
  v8h b0 = *(const v8h*)(bcol);
  v8h b1 = *(const v8h*)(bcol + 16);
#pragma unroll
  for (int q = 0; q < 8; ++q) {
    a[q] = a0[q]; a[8 + q] = a1[q];
    b[q] = b0[q]; b[8 + q] = b1[q];
  }
}

// M = Y^T Y rank-V update.  Y[v, c*51+j] = {u | w_shp | w_exp}[3v+c, j]; Y[v,153]=1.
__global__ __launch_bounds__(256)
void syrk_kernel(const float* __restrict__ u, const float* __restrict__ w_shp,
                 const float* __restrict__ w_exp, float* __restrict__ Mf) {
  __shared__ __align__(16) _Float16 Yt[2][KD * CHUNK];   // double-buffered slabs

  const int tid     = threadIdx.x;
  const int wave    = __builtin_amdgcn_readfirstlane(tid >> 5);  // scalar wave id
  const int lane    = tid & 31;
  const int lane16  = lane & 15;
  const int halfSel = lane >> 4;

  // per-wave tile list: tiles t = wave + 8*i over the 55 upper-triangle tiles
  int mts[7], nts[7];
#pragma unroll
  for (int i = 0; i < 7; ++i) {
    int t = wave + 8 * i;
    if (t >= NPAIRS) t = 0;            // dummy (recomputes tile 0, never written)
    int mt = 0, rem = t;
    while (rem >= NTILE - mt) { rem -= NTILE - mt; ++mt; }
    mts[i] = mt;
    nts[i] = mt + rem;
  }

  v8f zero = {};
  v8f acc[7];
#pragma unroll
  for (int i = 0; i < 7; ++i) acc[i] = zero;

  float rv[NLD];
  int chunk = blockIdx.x;                       // GRID < NCHUNKS: always valid
  load_regs(chunk * CHUNK, tid, u, w_shp, w_exp, rv);
  store_slab(Yt[0], tid, rv);
  __syncthreads();

  int cur = 0;
  for (;;) {
    const int nxt = chunk + GRID;
    const bool hasNext = nxt < NCHUNKS;         // uniform across workgroup
    if (hasNext) {
      load_regs(nxt * CHUNK, tid, u, w_shp, w_exp, rv);  // issue early, wait later
      // warm L2 for the chunk after that (gfx1250 global_prefetch)
      int pf = chunk + 2 * GRID;
      if (pf < NCHUNKS) {
        const char* p = (const char*)(w_shp + 3L * pf * CHUNK * 40);
        __builtin_prefetch(p + tid * 64, 0, 3);          // 16KB >= 15.4KB slab
        if (tid < 64) {
          const char* q = (const char*)(w_exp + 3L * pf * CHUNK * 10);
          __builtin_prefetch(q + tid * 64, 0, 3);        // 4KB >= 3.84KB slab
        }
      }
    }

    // WMMA pass over current slab, 2-deep pipelined tile-operand loads
    {
      const _Float16* buf = Yt[cur];
      v16h a, b, a2, b2;
      load_tile(buf, mts[0], nts[0], lane16, halfSel, a, b);
#pragma unroll
      for (int i = 0; i < 7; ++i) {
        if (i < 6) load_tile(buf, mts[i + 1], nts[i + 1], lane16, halfSel, a2, b2);
        acc[i] = __builtin_amdgcn_wmma_f32_16x16x32_f16(
            false, a, false, b, (short)0, acc[i], false, false);
        a = a2; b = b2;
      }
    }
    __syncthreads();                 // all waves done reading both slabs' last use
    if (!hasNext) break;
    store_slab(Yt[cur ^ 1], tid, rv);   // waits on the in-flight global loads here
    __syncthreads();                 // new slab visible to all waves
    cur ^= 1;
    chunk = nxt;
  }

  // C/D layout: VGPR r, lane l -> M = r + (l>=16 ? 8 : 0), N = l&15
  const int Moff = halfSel * 8;
#pragma unroll
  for (int i = 0; i < 7; ++i) {
    int t = wave + 8 * i;
    if (t < NPAIRS) {
      int rowBase = mts[i] * 16 + Moff;
      int colIdx  = nts[i] * 16 + lane16;
#pragma unroll
      for (int r = 0; r < 8; ++r)
        atomicAdd(&Mf[(rowBase + r) * KD + colIdx], acc[i][r]);
    }
  }
}

__global__ __launch_bounds__(64)
void finish_kernel(const float* __restrict__ ws, float* __restrict__ out) {
  const int n   = blockIdx.x;
  const int tid = threadIdx.x;   // 64 threads
  const float* Mf    = ws + WS_M;
  const float* alpha = ws + WS_ALPHA + n * 64;
  const float* avec  = ws + WS_AVEC + n * 144;
  const float* sqd   = ws + WS_SQD + n * 12;

  __shared__ float al[51];
  __shared__ float G[16];      // upper-triangle of 4x4 Gram-of-basis form
  __shared__ float wts[12];
  if (tid < 51) al[tid] = alpha[tid];
  if (tid < 16) G[tid] = 0.0f;
  __syncthreads();

  const int CA[6] = {0, 0, 0, 1, 1, 2};
  const int CB[6] = {0, 1, 2, 1, 2, 2};
  // G[c][c'] = alpha^T M_{cc'} alpha ; G[c][3] = alpha . colsum_c ; G[3][3] = V
  for (int t = tid; t < 459; t += 64) {
    if (t < 306) {
      int p = t / 51, k = t % 51;
      int ca = CA[p], cb = CB[p];
      int r = ca * 51 + k;
      float s = 0.0f;
      for (int k2 = 0; k2 < 51; ++k2)
        s += al[k2] * getM(Mf, r, cb * 51 + k2);
      atomicAdd(&G[ca * 4 + cb], al[k] * s);
    } else {
      int t2 = t - 306;
      int c = t2 / 51, k = t2 % 51;
      atomicAdd(&G[c * 4 + 3], al[k] * getM(Mf, c * 51 + k, 153));
    }
  }
  if (tid == 0) G[15] = getM(Mf, 153, 153);   // = V (from ones column)
  __syncthreads();

  if (tid < 12) {
    int k = tid;
    float w2 = 0.0f;
    for (int r = 0; r < 3; ++r) {
      const float* a = &avec[k * 12 + r * 4];
      for (int q = 0; q < 4; ++q)
        for (int q2 = 0; q2 < 4; ++q2) {
          float g = (q <= q2) ? G[q * 4 + q2] : G[q2 * 4 + q];
          w2 += a[q] * g * a[q2];
        }
    }
    float w = sqrtf(fmaxf(w2, 0.0f));
    if (k < 11) w += 1e-6f;
    wts[k] = w;
  }
  __syncthreads();
  if (tid == 0) {
    float mx = wts[0];
    for (int k = 1; k < 12; ++k) mx = fmaxf(mx, wts[k]);
    float s = 0.0f;
    for (int k = 0; k < 12; ++k) s += (wts[k] / mx) * sqd[k];
    atomicAdd(out, s * (1.0f / (NB * 12.0f)));
  }
}

extern "C" void kernel_launch(void* const* d_in, const int* in_sizes, int n_in,
                              void* d_out, int out_size, void* d_ws, size_t ws_size,
                              hipStream_t stream) {
  (void)in_sizes; (void)n_in; (void)out_size; (void)ws_size;
  const float* inp    = (const float*)d_in[0];
  const float* target = (const float*)d_in[1];
  const float* pmean  = (const float*)d_in[2];
  const float* pstd   = (const float*)d_in[3];
  const float* u      = (const float*)d_in[4];
  const float* w_shp  = (const float*)d_in[5];
  const float* w_exp  = (const float*)d_in[6];
  float* out = (float*)d_out;
  float* ws  = (float*)d_ws;

  zero_kernel<<<(KD * KD + 255) / 256, 256, 0, stream>>>(ws + WS_M, out);
  prep_kernel<<<1, 256, 0, stream>>>(inp, target, pmean, pstd, ws);
  syrk_kernel<<<GRID, 256, 0, stream>>>(u, w_shp, w_exp, ws + WS_M);
  finish_kernel<<<NB, 64, 0, stream>>>(ws, out);
}
